// POSTransformer_50629074485674
// MI455X (gfx1250) — compile-verified, matrix operand order checked by reading, not run on hardware
//
#include <hip/hip_runtime.h>
#include <cstdint>
#include <cstddef>

// ---------------------------------------------------------------------------
// POS transformer forward for MI455X (gfx1250, wave32, WMMA).
// Compute-bound (~206 GFLOP vs <100MB @ 23.3TB/s) -> everything runs on
// v_wmma_f32_16x16x32_f16.  Weights/activations converted to f16 ONCE per
// launch; GEMM hot loop is pure async-copy + ds_load + wmma (no cvt).
// ---------------------------------------------------------------------------

#define L_LAYERS 6
#define HNUM     8
#define DMODEL   512
#define SEQ      1024
#define BATCH    4
#define FFN      2048
#define HD       64
#define PDIM     64
#define TOKENS   (BATCH * SEQ)   // 4096
#define EPS      1e-5f

typedef __attribute__((ext_vector_type(16))) _Float16 v16h;
typedef __attribute__((ext_vector_type(8)))  _Float16 v8h;
typedef __attribute__((ext_vector_type(4)))  _Float16 v4h;
typedef __attribute__((ext_vector_type(8)))  float    v8f;
typedef __attribute__((ext_vector_type(4)))  int      v4i;

// WMMA: D = A(16x32 f16) * B(32x16 f16) + C(16x16 f32)
__device__ __forceinline__ v8f wmma_f32_f16(v16h a, v16h b, v8f c) {
  return __builtin_amdgcn_wmma_f32_16x16x32_f16(
      false, a, false, b, (short)0, c, false, false);
}

// ---------------------------------------------------------------------------
// gfx1250 async global->LDS copy (ASYNCcnt-tracked), with sync fallback.
// Signature (from clang diagnostic): param0 = v4i addrspace(1)*, and per
// LLVM convention param1 = v4i addrspace(3)*, then imm offset + imm cpol.
// ---------------------------------------------------------------------------
#if defined(__gfx1250__) && __has_builtin(__builtin_amdgcn_global_load_async_to_lds_b128) && __has_builtin(__builtin_amdgcn_s_wait_asynccnt)
#define USE_ASYNC_LDS 1
#else
#define USE_ASYNC_LDS 0
#endif

__device__ __forceinline__ void copy16B_g2l(const _Float16* g, _Float16* l) {
#if USE_ASYNC_LDS
  __builtin_amdgcn_global_load_async_to_lds_b128(
      (__attribute__((address_space(1))) v4i*)g,
      (__attribute__((address_space(3))) v4i*)l, 0, 0);
#else
  *(v8h*)l = *(const v8h*)g;
#endif
}

__device__ __forceinline__ void wait_stage() {
#if USE_ASYNC_LDS
  __builtin_amdgcn_s_wait_asynccnt(0);
#endif
  __syncthreads();
}

// ---------------------------------------------------------------------------
// Fragment loaders.  16-bit 16x32 A/B layout (ISA 7.12.2): lane&15 = row
// (A) / col (B); kb = (lane>>4)*8; elems 0..7 = K kb..kb+7, 8..15 = K+16.
// ---------------------------------------------------------------------------
__device__ __forceinline__ v16h frag_lds(const _Float16* base, int stride) {
  int lane = threadIdx.x & 31;
  const _Float16* p = base + (lane & 15) * stride + ((lane >> 4) * 8);
  v8h lo = *(const v8h*)(p);
  v8h hi = *(const v8h*)(p + 16);
  v16h f;
#pragma unroll
  for (int i = 0; i < 8; ++i) { f[i] = lo[i]; f[8 + i] = hi[i]; }
  return f;
}

__device__ __forceinline__ v16h frag_global_f16(const _Float16* base, int rstride) {
  int lane = threadIdx.x & 31;
  const _Float16* p = base + (size_t)(lane & 15) * rstride + ((lane >> 4) * 8);
  v8h lo = *(const v8h*)(p);
  v8h hi = *(const v8h*)(p + 16);
  v16h f;
#pragma unroll
  for (int i = 0; i < 8; ++i) { f[i] = lo[i]; f[8 + i] = hi[i]; }
  return f;
}

// ---------------------------------------------------------------------------
// f32 -> f16 bulk convert (weights once per launch; x before head GEMM)
// ---------------------------------------------------------------------------
__global__ void cvt_f16_kernel(const float* __restrict__ src,
                               _Float16* __restrict__ dst, int n) {
  int i = (blockIdx.x * blockDim.x + threadIdx.x) * 8;
  if (i >= n) return;
  float4 a = *(const float4*)(src + i);
  float4 b = *(const float4*)(src + i + 4);
  v8h h = {(_Float16)a.x, (_Float16)a.y, (_Float16)a.z, (_Float16)a.w,
           (_Float16)b.x, (_Float16)b.y, (_Float16)b.z, (_Float16)b.w};
  *(v8h*)(dst + i) = h;
}

// ---------------------------------------------------------------------------
// Embedding gather: x[t,:] = emb[tokens[t],:]  (residual stream stays f32)
// ---------------------------------------------------------------------------
__global__ void embed_kernel(const int* __restrict__ tokens,
                             const float* __restrict__ emb,
                             float* __restrict__ x) {
  int t = blockIdx.x;
  int tok = tokens[t];
  const float4* src = (const float4*)(emb + (size_t)tok * DMODEL);
  float4* dst = (float4*)(x + (size_t)t * DMODEL);
  for (int i = threadIdx.x; i < DMODEL / 4; i += blockDim.x) dst[i] = src[i];
}

// ---------------------------------------------------------------------------
// LayerNorm: one wave32 per row of 512, shfl_xor reductions, f16 output.
// ---------------------------------------------------------------------------
__global__ void ln_kernel(const float* __restrict__ x,
                          const float* __restrict__ w,
                          const float* __restrict__ b,
                          _Float16* __restrict__ y) {
  int lane = threadIdx.x & 31;
  int wave = threadIdx.x >> 5;
  int row  = blockIdx.x * (blockDim.x >> 5) + wave;
  const float* xr = x + (size_t)row * DMODEL;

  float vals[16];
  float s = 0.f;
#pragma unroll
  for (int t = 0; t < 4; ++t) {
    float4 f = *(const float4*)(xr + lane * 4 + t * 128);
    vals[t * 4 + 0] = f.x; vals[t * 4 + 1] = f.y;
    vals[t * 4 + 2] = f.z; vals[t * 4 + 3] = f.w;
    s += f.x + f.y + f.z + f.w;
  }
#pragma unroll
  for (int m = 1; m < 32; m <<= 1) s += __shfl_xor(s, m, 32);
  float mu = s * (1.0f / DMODEL);

  float vs = 0.f;
#pragma unroll
  for (int i = 0; i < 16; ++i) { float d = vals[i] - mu; vs += d * d; }
#pragma unroll
  for (int m = 1; m < 32; m <<= 1) vs += __shfl_xor(vs, m, 32);
  float rs = rsqrtf(vs * (1.0f / DMODEL) + EPS);

  _Float16* yr = y + (size_t)row * DMODEL;
#pragma unroll
  for (int t = 0; t < 4; ++t) {
    int idx = lane * 4 + t * 128;
    float4 wv = *(const float4*)(w + idx);
    float4 bv = *(const float4*)(b + idx);
    v4h o = {(_Float16)((vals[t * 4 + 0] - mu) * rs * wv.x + bv.x),
             (_Float16)((vals[t * 4 + 1] - mu) * rs * wv.y + bv.y),
             (_Float16)((vals[t * 4 + 2] - mu) * rs * wv.z + bv.z),
             (_Float16)((vals[t * 4 + 3] - mu) * rs * wv.w + bv.w)};
    *(v4h*)(yr + idx) = o;
  }
}

// ---------------------------------------------------------------------------
// GEMM: C[M,N] = A[M,K](f16) * W[N,K](f16)^T + bias[N]
// MODE 0: +bias    MODE 1: +bias+residual(f32)    MODE 2: +bias, exact GELU
// 256 threads / 8 waves; block tile 128x64, BK=32; wave tile 32x32 (4 WMMA).
// Double-buffered LDS, async global->LDS copies overlap WMMA of prev tile.
// ---------------------------------------------------------------------------
#define LSTR 40   // f16 LDS row stride (80B: 16B-aligned, conflict-free)

template <int MODE, typename OutT>
__launch_bounds__(256)
__global__ void gemm_kernel(const _Float16* __restrict__ A,
                            const _Float16* __restrict__ W,
                            const float* __restrict__ bias,
                            const float* __restrict__ res,
                            OutT* __restrict__ C,
                            int M, int N, int K) {
  __shared__ __align__(16) _Float16 ldsA[2][128 * LSTR];
  __shared__ __align__(16) _Float16 ldsB[2][64 * LSTR];

  int tid  = threadIdx.x;
  int lane = tid & 31;
  int wave = tid >> 5;
  int m0 = blockIdx.y * 128;
  int n0 = blockIdx.x * 64;
  int wm = wave >> 1;      // 0..3 : 32-row strip
  int wn = wave & 1;       // 0..1 : 32-col strip

  v8f acc[2][2] = {{{}, {}}, {{}, {}}};

  int rowA = tid >> 1;            // 0..127
  int colA = (tid & 1) * 16;      // 0,16
  const _Float16* aBase = A + (size_t)(m0 + rowA) * K + colA;
  _Float16* aDst0 = &ldsA[0][rowA * LSTR + colA];
  _Float16* aDst1 = &ldsA[1][rowA * LSTR + colA];

  int rowB = (tid & 127) >> 1;    // 0..63 (waves 0..3 only)
  int colB = (tid & 1) * 16;
  const _Float16* bBase = W + (size_t)(n0 + rowB) * K + colB;
  _Float16* bDst0 = &ldsB[0][rowB * LSTR + colB];
  _Float16* bDst1 = &ldsB[1][rowB * LSTR + colB];

  int nsteps = K / 32;

  // prologue: stage tile 0 into buffer 0
  copy16B_g2l(aBase, aDst0);
  copy16B_g2l(aBase + 8, aDst0 + 8);
  if (tid < 128) {
    copy16B_g2l(bBase, bDst0);
    copy16B_g2l(bBase + 8, bDst0 + 8);
  }

  for (int k = 0; k < nsteps; ++k) {
    wait_stage();   // tile k resident + all waves past previous reads

    // stage tile k+1 into the other buffer while we compute on tile k
    if (k + 1 < nsteps) {
      const _Float16* ag = aBase + (k + 1) * 32;
      _Float16* ad = ((k + 1) & 1) ? aDst1 : aDst0;
      copy16B_g2l(ag, ad);
      copy16B_g2l(ag + 8, ad + 8);
      if (tid < 128) {
        const _Float16* bg = bBase + (k + 1) * 32;
        _Float16* bd = ((k + 1) & 1) ? bDst1 : bDst0;
        copy16B_g2l(bg, bd);
        copy16B_g2l(bg + 8, bd + 8);
      }
    }

    const _Float16* ldA = ldsA[k & 1];
    const _Float16* ldB = ldsB[k & 1];
    v16h a0 = frag_lds(ldA + (wm * 32 + 0)  * LSTR, LSTR);
    v16h a1 = frag_lds(ldA + (wm * 32 + 16) * LSTR, LSTR);
    v16h b0 = frag_lds(ldB + (wn * 32 + 0)  * LSTR, LSTR);
    v16h b1 = frag_lds(ldB + (wn * 32 + 16) * LSTR, LSTR);
    acc[0][0] = wmma_f32_f16(a0, b0, acc[0][0]);
    acc[0][1] = wmma_f32_f16(a0, b1, acc[0][1]);
    acc[1][0] = wmma_f32_f16(a1, b0, acc[1][0]);
    acc[1][1] = wmma_f32_f16(a1, b1, acc[1][1]);
  }

  // epilogue: C layout = (M = v + 8*(lane>=16), N = lane&15)
  int n  = lane & 15;
  int mb = (lane >> 4) * 8;
#pragma unroll
  for (int i = 0; i < 2; ++i) {
#pragma unroll
    for (int j = 0; j < 2; ++j) {
      int gn = n0 + wn * 32 + j * 16 + n;
      float bv = bias[gn];
#pragma unroll
      for (int v = 0; v < 8; ++v) {
        size_t idx = (size_t)(m0 + wm * 32 + i * 16 + mb + v) * N + gn;
        float val = acc[i][j][v] + bv;
        if (MODE == 1) val += res[idx];
        if (MODE == 2) val = 0.5f * val * (1.0f + erff(val * 0.70710678118f));
        C[idx] = (OutT)val;
      }
    }
  }
}

// ---------------------------------------------------------------------------
// Flash attention with ALiBi.  128 threads = 4 waves; block owns 64 queries
// of one (b,h); waves share K/V^T tiles staged in LDS.  Per 32-key block:
// 4 score WMMAs + 4 PV WMMAs per wave; online softmax on 16-lane row groups.
// ---------------------------------------------------------------------------
__launch_bounds__(128)
__global__ void attn_kernel(const _Float16* __restrict__ qkvh,
                            _Float16* __restrict__ attn_out) {
  __shared__ __align__(16) _Float16 ldsK[32 * 72];        // [key][dim0..63]
  __shared__ __align__(16) _Float16 ldsVT[64 * LSTR];     // [dim][key0..31]
  __shared__ __align__(16) _Float16 ldsP[4][16 * LSTR];   // per-wave probs

  int tid  = threadIdx.x;
  int lane = tid & 31;
  int wave = tid >> 5;
  int h = blockIdx.y;
  int b = blockIdx.z;
  int qs = blockIdx.x * 64 + wave * 16;   // this wave's 16-query tile
  float slope = exp2f(-8.0f / (float)(h + 1));
  const float scale = 0.125f;             // 1/sqrt(64)
  const size_t rstr = 3 * DMODEL;

  const _Float16* qbase = qkvh + (size_t)(b * SEQ + qs) * rstr + h * HD;
  v16h q0 = frag_global_f16(qbase, rstr);
  v16h q1 = frag_global_f16(qbase + 32, rstr);

  v8f o[4] = {{}, {}, {}, {}};
  float mrow[8], lrow[8];
#pragma unroll
  for (int v = 0; v < 8; ++v) { mrow[v] = -1e30f; lrow[v] = 0.f; }

  int n  = lane & 15;
  int mb = (lane >> 4) * 8;
  int skey  = tid >> 2;            // 0..31  staging: key row
  int schunk = (tid & 3) * 16;     // 0,16,32,48 : dim chunk

  for (int j = 0; j < SEQ / 32; ++j) {
    __syncthreads();   // previous tile fully consumed

    // ---- stage K tile [32 keys x 64 dims] (async path) ----
    const _Float16* kg =
        qkvh + (size_t)(b * SEQ + j * 32 + skey) * rstr + DMODEL + h * HD + schunk;
    copy16B_g2l(kg,     &ldsK[skey * 72 + schunk]);
    copy16B_g2l(kg + 8, &ldsK[skey * 72 + schunk + 8]);

    // ---- stage V^T tile [64 dims x 32 keys] (transpose via registers) ----
    const _Float16* vg =
        qkvh + (size_t)(b * SEQ + j * 32 + skey) * rstr + 2 * DMODEL + h * HD + schunk;
    v8h vlo = *(const v8h*)(vg);
    v8h vhi = *(const v8h*)(vg + 8);
#pragma unroll
    for (int i = 0; i < 8; ++i) {
      ldsVT[(schunk + i) * LSTR + skey]     = vlo[i];
      ldsVT[(schunk + 8 + i) * LSTR + skey] = vhi[i];
    }
    wait_stage();

    // ---- scores: two 16x16 tiles over 32 keys, K-dim 64 in two chunks ----
    v8f z = {};
    v8f c0 = wmma_f32_f16(q0, frag_lds(&ldsK[0],           72), z);
    c0     = wmma_f32_f16(q1, frag_lds(&ldsK[32],          72), c0);
    v8f c1 = wmma_f32_f16(q0, frag_lds(&ldsK[16 * 72],     72), z);
    c1     = wmma_f32_f16(q1, frag_lds(&ldsK[16 * 72 + 32], 72), c1);

#pragma unroll
    for (int v = 0; v < 8; ++v) {
      float qi = (float)(qs + mb + v);
      c0[v] = c0[v] * scale + slope * ((float)(j * 32 + n) - qi);
      c1[v] = c1[v] * scale + slope * ((float)(j * 32 + 16 + n) - qi);
    }

    // ---- online softmax (row stats on 16-lane half-groups) ----
    float p0[8], p1[8], alpha[8];
#pragma unroll
    for (int v = 0; v < 8; ++v) {
      float rmax = fmaxf(c0[v], c1[v]);
#pragma unroll
      for (int m = 1; m < 16; m <<= 1) rmax = fmaxf(rmax, __shfl_xor(rmax, m, 32));
      float mnew = fmaxf(mrow[v], rmax);
      alpha[v] = __expf(mrow[v] - mnew);
      p0[v] = __expf(c0[v] - mnew);
      p1[v] = __expf(c1[v] - mnew);
      float rsum = p0[v] + p1[v];
#pragma unroll
      for (int m = 1; m < 16; m <<= 1) rsum += __shfl_xor(rsum, m, 32);
      lrow[v] = lrow[v] * alpha[v] + rsum;
      mrow[v] = mnew;
    }
#pragma unroll
    for (int t = 0; t < 4; ++t)
#pragma unroll
      for (int v = 0; v < 8; ++v) o[t][v] *= alpha[v];

    // ---- P (C-layout) -> A-fragment layout via this wave's LDS slab ----
    _Float16* pw = &ldsP[wave][0];
#pragma unroll
    for (int v = 0; v < 8; ++v) {
      pw[(mb + v) * LSTR + n]      = (_Float16)p0[v];
      pw[(mb + v) * LSTR + 16 + n] = (_Float16)p1[v];
    }
    v16h pa = frag_lds(pw, LSTR);

    // ---- O += P * V : B-frag cols = head dims, K = keys (ldsVT rows) ----
#pragma unroll
    for (int t = 0; t < 4; ++t)
      o[t] = wmma_f32_f16(pa, frag_lds(&ldsVT[(t * 16) * LSTR], LSTR), o[t]);
  }

  // ---- normalize, write f16 out [T, D] at columns h*64.. ----
#pragma unroll
  for (int v = 0; v < 8; ++v) {
    float il = 1.0f / lrow[v];
    size_t row = (size_t)(b * SEQ + qs + mb + v);
#pragma unroll
    for (int t = 0; t < 4; ++t)
      attn_out[row * DMODEL + h * HD + t * 16 + n] = (_Float16)(o[t][v] * il);
  }
}

// ---------------------------------------------------------------------------
// Host orchestration (all on `stream`, graph-capture safe).
// ---------------------------------------------------------------------------
static inline int cvt_grid(int n) { return (n / 8 + 255) / 256; }

extern "C" void kernel_launch(void* const* d_in, const int* in_sizes, int n_in,
                              void* d_out, int out_size, void* d_ws, size_t ws_size,
                              hipStream_t stream) {
  (void)in_sizes; (void)n_in; (void)out_size; (void)ws_size;

  const int*   tokens = (const int*)  d_in[0];
  const float* emb    = (const float*)d_in[1];
  const float* wqkv   = (const float*)d_in[2];
  const float* bqkv   = (const float*)d_in[3];
  const float* wo     = (const float*)d_in[4];
  const float* bo     = (const float*)d_in[5];
  const float* ln1w   = (const float*)d_in[6];
  const float* ln1b   = (const float*)d_in[7];
  const float* ln2w   = (const float*)d_in[8];
  const float* ln2b   = (const float*)d_in[9];
  const float* w1     = (const float*)d_in[10];
  const float* b1     = (const float*)d_in[11];
  const float* w2     = (const float*)d_in[12];
  const float* b2     = (const float*)d_in[13];
  const float* headw  = (const float*)d_in[14];
  const float* headb  = (const float*)d_in[15];
  float* out = (float*)d_out;

  // ---- workspace carve-up ----
  const size_t nX    = (size_t)TOKENS * DMODEL;       // f32 residual
  const size_t nXH   = (size_t)TOKENS * DMODEL;       // f16
  const size_t nQKV  = (size_t)TOKENS * 3 * DMODEL;   // f16
  const size_t nATT  = (size_t)TOKENS * DMODEL;       // f16
  const size_t nHB   = (size_t)TOKENS * FFN;          // f16
  const size_t nWQKV = (size_t)L_LAYERS * 3 * DMODEL * DMODEL;
  const size_t nWO   = (size_t)L_LAYERS * DMODEL * DMODEL;
  const size_t nW1   = (size_t)L_LAYERS * FFN * DMODEL;
  const size_t nW2   = (size_t)L_LAYERS * DMODEL * FFN;
  const size_t nHW   = (size_t)PDIM * DMODEL;

  float* x = (float*)d_ws;
  _Float16* p = (_Float16*)(x + nX);
  _Float16* xh    = p; p += nXH;
  _Float16* qkvh  = p; p += nQKV;
  _Float16* atth  = p; p += nATT;
  _Float16* hbh   = p; p += nHB;
  _Float16* wqkvh = p; p += nWQKV;
  _Float16* woh   = p; p += nWO;
  _Float16* w1h   = p; p += nW1;
  _Float16* w2h   = p; p += nW2;
  _Float16* headh = p; p += nHW;

  // ---- one-time f32 -> f16 weight conversion ----
  cvt_f16_kernel<<<cvt_grid((int)nWQKV), 256, 0, stream>>>(wqkv, wqkvh, (int)nWQKV);
  cvt_f16_kernel<<<cvt_grid((int)nWO),   256, 0, stream>>>(wo,   woh,   (int)nWO);
  cvt_f16_kernel<<<cvt_grid((int)nW1),   256, 0, stream>>>(w1,   w1h,   (int)nW1);
  cvt_f16_kernel<<<cvt_grid((int)nW2),   256, 0, stream>>>(w2,   w2h,   (int)nW2);
  cvt_f16_kernel<<<cvt_grid((int)nHW),   256, 0, stream>>>(headw, headh, (int)nHW);

  embed_kernel<<<TOKENS, 128, 0, stream>>>(tokens, emb, x);

  for (int l = 0; l < L_LAYERS; ++l) {
    ln_kernel<<<TOKENS / 8, 256, 0, stream>>>(x, ln1w + l * DMODEL, ln1b + l * DMODEL, xh);

    gemm_kernel<0, _Float16><<<dim3(3 * DMODEL / 64, TOKENS / 128), 256, 0, stream>>>(
        xh, wqkvh + (size_t)l * 3 * DMODEL * DMODEL,
        bqkv + (size_t)l * 3 * DMODEL, nullptr, qkvh,
        TOKENS, 3 * DMODEL, DMODEL);

    attn_kernel<<<dim3(SEQ / 64, HNUM, BATCH), 128, 0, stream>>>(qkvh, atth);

    gemm_kernel<1, float><<<dim3(DMODEL / 64, TOKENS / 128), 256, 0, stream>>>(
        atth, woh + (size_t)l * DMODEL * DMODEL,
        bo + (size_t)l * DMODEL, x, x,
        TOKENS, DMODEL, DMODEL);

    ln_kernel<<<TOKENS / 8, 256, 0, stream>>>(x, ln2w + l * DMODEL, ln2b + l * DMODEL, xh);

    gemm_kernel<2, _Float16><<<dim3(FFN / 64, TOKENS / 128), 256, 0, stream>>>(
        xh, w1h + (size_t)l * FFN * DMODEL,
        b1 + (size_t)l * FFN, nullptr, hbh,
        TOKENS, FFN, DMODEL);

    gemm_kernel<1, float><<<dim3(DMODEL / 64, TOKENS / 128), 256, 0, stream>>>(
        hbh, w2h + (size_t)l * DMODEL * FFN,
        b2 + (size_t)l * DMODEL, x, x,
        TOKENS, DMODEL, FFN);
  }

  // final projection: convert residual stream once, then f16 GEMM -> f32 out
  cvt_f16_kernel<<<cvt_grid((int)nX), 256, 0, stream>>>(x, xh, (int)nX);
  gemm_kernel<0, float><<<dim3(PDIM / 64, TOKENS / 128), 256, 0, stream>>>(
      xh, headh, headb, nullptr, out, TOKENS, PDIM, DMODEL);
}
